// DW_Attention_11879879544365
// MI455X (gfx1250) — matrix operand order repaired and checked by reading, not compiled, not run on hardware
//
#include <hip/hip_runtime.h>

// ---------------- CDNA5 WMMA types / helpers ----------------
typedef _Float16 h8   __attribute__((ext_vector_type(8)));
typedef _Float16 v16h __attribute__((ext_vector_type(16)));
typedef float    v8f  __attribute__((ext_vector_type(8)));

// A-matrix fragment (16x32 f16, MxK): lane = M (lanes 0-15 and 16-31 both map M=0..15),
// element e -> K = (e<8 ? 0 : 16) + 8*(lane>>4) + (e&7).  Caller passes p = row + k0 + 8*half.
__device__ __forceinline__ v16h frag_a(const _Float16* p) {
  h8 lo = *(const h8*)(p);
  h8 hi = *(const h8*)(p + 16);
  return __builtin_shufflevector(lo, hi, 0,1,2,3,4,5,6,7,8,9,10,11,12,13,14,15);
}
// B-matrix fragment (32x16 f16, KxN): lane = N, element e -> K = (e<8 ? 0 : 16) + 8*(lane>>4) + (e&7).
// Caller passes p = row + k0 + 8*half.
__device__ __forceinline__ v16h frag_b(const _Float16* p) {
  h8 lo = *(const h8*)(p);
  h8 hi = *(const h8*)(p + 16);
  return __builtin_shufflevector(lo, hi, 0,1,2,3,4,5,6,7,8,9,10,11,12,13,14,15);
}
__device__ __forceinline__ v8f wmma16(v16h a, v16h b, v8f c) {
  return __builtin_amdgcn_wmma_f32_16x16x32_f16(false, a, false, b, (short)0, c, false, false);
}

// token s (s = B_idx*256 + HW_idx) -> row in x layout (b,l,hx,wy,w1,w2)
__device__ __forceinline__ int xrow_of(int s) {
  return (s & 0xC000) | (((s >> 6) & 3) << 12) | (((s >> 11) & 7) << 9)
       | (((s >> 8) & 7) << 6) | (s & 63);
}

// ---------------- weight f32 -> f16 conversion ----------------
__global__ __launch_bounds__(256) void cvt_w_f16(const float* __restrict__ w,
                                                 _Float16* __restrict__ o, int n) {
  int i = blockIdx.x * 256 + threadIdx.x;
  if (i < n) o[i] = (_Float16)w[i];
}

// ---------------- fused window attention ----------------
// One block (8 wave32) per window.  GROUP selects channel slice [GROUP*128, GROUP*128+128).
template <int WS, int GROUP>
__global__ __launch_bounds__(256) void win_attn(
    const float* __restrict__ x, const _Float16* __restrict__ w16,
    const float* __restrict__ qkv_b, const float* __restrict__ table,
    float* __restrict__ out) {
  constexpr int N   = WS * WS;          // tokens per window (16 or 64)
  constexpr int LOG = (WS == 4) ? 2 : 3;
  constexpr int MT  = N / 16;           // M tiles (1 or 4)
  constexpr int NP  = (WS == 4) ? 32 : 64; // K dim (zero padded) for P@V
  constexpr int PERS = 3 * N * 128;     // halfs: persistent q,k,v
  constexpr int OVL  = N * 256;         // halfs: overlay (sX, later sS+sP+sVt)

  __shared__ __align__(16) _Float16 smem[PERS + OVL];
  _Float16* sQ  = smem;
  _Float16* sK  = sQ + N * 128;
  _Float16* sV  = sK + N * 128;
  _Float16* sX  = smem + PERS;                       // phase 0-1
  float*    sS  = (float*)(smem + PERS);             // phase 2 (reuses sX)
  _Float16* sP  = smem + PERS + 2 * N * N;           // after N*N floats
  _Float16* sVt = sP + N * NP;

  const int tid  = threadIdx.x;
  const int wv   = tid >> 5;
  const int ln   = tid & 31;
  const int l16  = ln & 15;
  const int half = ln >> 4;

  int wid = blockIdx.x, lb, wh, ww;
  if (WS == 4) { lb = wid >> 8; wh = (wid >> 4) & 15; ww = wid & 15; }
  else         { lb = wid >> 6; wh = (wid >> 3) & 7;  ww = wid & 7;  }
  const int sBase = (lb << 12) + ((wh * WS) << 6) + ww * WS;

  // ---- Phase 0: stage window token rows, f32 -> f16 ----
  for (int i = wv; i < N; i += 8) {
    int s = sBase + ((i >> LOG) << 6) + (i & (WS - 1));
    const float4* src = (const float4*)(x + (size_t)xrow_of(s) * 256) + ln * 2;
    float4 f0 = src[0], f1 = src[1];
    h8 v;
    v[0] = (_Float16)f0.x; v[1] = (_Float16)f0.y; v[2] = (_Float16)f0.z; v[3] = (_Float16)f0.w;
    v[4] = (_Float16)f1.x; v[5] = (_Float16)f1.y; v[6] = (_Float16)f1.z; v[7] = (_Float16)f1.w;
    *(h8*)(sX + i * 256 + ln * 8) = v;
  }
  __syncthreads();

  // ---- Phase 1: fused QKV GEMM (M=N tokens, Nout=384, K=256) ----
  for (int t = wv; t < MT * 24; t += 8) {
    int mT = t / 24, nT = t % 24;
    int nl  = nT * 16 + l16;          // 0..383
    int sec = nl >> 7;                // 0=q 1=k 2=v
    int cI  = nl & 127;
    int wr  = sec * 256 + GROUP * 128 + cI;  // row in qkv_w
    const _Float16* ap = sX  + (mT * 16 + l16) * 256 + half * 8;
    const _Float16* bp = w16 + wr * 256 + half * 8;
    v8f acc = {0.f, 0.f, 0.f, 0.f, 0.f, 0.f, 0.f, 0.f};
#pragma unroll
    for (int k0 = 0; k0 < 256; k0 += 32)
      acc = wmma16(frag_a(ap + k0), frag_b(bp + k0), acc);
    float bias  = qkv_b[wr];
    float scale = (sec == 0) ? 0.17677669529663687f : 1.0f;  // gc^-0.5 on q
    _Float16* dst = (sec == 0) ? sQ : (sec == 1) ? sK : sV;
#pragma unroll
    for (int r = 0; r < 8; ++r) {
      int tok = mT * 16 + r + half * 8;
      dst[tok * 128 + cI] = (_Float16)((acc[r] + bias) * scale);
    }
  }
  __syncthreads();

  // ---- Phase 2: per-head attention ----
  for (int h = 0; h < 4; ++h) {
    // S = Q K^T + rel-pos bias  (gc=32 -> single WMMA per 16x16 tile)
    for (int t = wv; t < MT * MT; t += 8) {
      int iT = t / MT, jT = t % MT;
      v16h a = frag_a(sQ + (iT * 16 + l16) * 128 + h * 32 + half * 8);
      v16h b = frag_b(sK + (jT * 16 + l16) * 128 + h * 32 + half * 8);
      v8f d = {0.f, 0.f, 0.f, 0.f, 0.f, 0.f, 0.f, 0.f};
      d = wmma16(a, b, d);
      int j = jT * 16 + l16, yj = j >> LOG, xj = j & (WS - 1);
#pragma unroll
      for (int r = 0; r < 8; ++r) {
        int i = iT * 16 + r + half * 8;
        int yi = i >> LOG, xi = i & (WS - 1);
        int idx = (yi - yj + WS - 1) * (2 * WS - 1) + (xi - xj + WS - 1);
        sS[i * N + j] = d[r] + table[idx * 4 + h];
      }
    }
    __syncthreads();

    // stage V^T for this head (zero pad K to NP)
    for (int t = tid; t < 32 * NP; t += 256) {
      int c = t / NP, k = t % NP;
      sVt[c * NP + k] = (k < N) ? sV[k * 128 + h * 32 + c] : (_Float16)0.0f;
    }
    // softmax: one thread per row
    if (tid < N) {
      float mx = -3.0e38f;
      for (int j = 0; j < N; ++j) mx = fmaxf(mx, sS[tid * N + j]);
      float sum = 0.f;
      for (int j = 0; j < N; ++j) {
        float e = __expf(sS[tid * N + j] - mx);
        sum += e;
        sS[tid * N + j] = e;
      }
      float inv = 1.0f / sum;
      for (int j = 0; j < NP; ++j)
        sP[tid * NP + j] = (j < N) ? (_Float16)(sS[tid * N + j] * inv) : (_Float16)0.0f;
    }
    __syncthreads();

    // O = P V  (NxN @ Nx32), stores go straight to the permuted output layout
    for (int t = wv; t < MT * 2; t += 8) {
      int mT = t >> 1, nT = t & 1;
      v8f acc = {0.f, 0.f, 0.f, 0.f, 0.f, 0.f, 0.f, 0.f};
#pragma unroll
      for (int k0 = 0; k0 < NP; k0 += 32)
        acc = wmma16(frag_a(sP  + (mT * 16 + l16) * NP + k0 + half * 8),
                     frag_b(sVt + (nT * 16 + l16) * NP + k0 + half * 8), acc);
      int chan = GROUP * 128 + h * 32 + nT * 16 + l16;       // 0..255
      int orbase = ((chan >> 6) << 12) + (((chan >> 3) & 7) << 3) + (chan & 7);
#pragma unroll
      for (int r = 0; r < 8; ++r) {
        int i = mT * 16 + r + half * 8;                       // token in window
        int s = sBase + ((i >> LOG) << 6) + (i & (WS - 1));   // global token id
        int orow = (s & 0xC000) | (((s >> 11) & 7) << 9) | (((s >> 8) & 7) << 6);
        out[(size_t)(orow + orbase) * 256 + (s & 255)] = acc[r];
      }
    }
    __syncthreads();
  }
}

// ---------------- host launcher ----------------
extern "C" void kernel_launch(void* const* d_in, const int* in_sizes, int n_in,
                              void* d_out, int out_size, void* d_ws, size_t ws_size,
                              hipStream_t stream) {
  const float* x     = (const float*)d_in[0];
  const float* qkv_w = (const float*)d_in[1];
  const float* qkv_b = (const float*)d_in[2];
  const float* tab0  = (const float*)d_in[3];
  const float* tab1  = (const float*)d_in[4];
  float* out = (float*)d_out;

  _Float16* w16 = (_Float16*)d_ws;     // 768*256 f16 = 384 KB
  const int nw = 768 * 256;
  cvt_w_f16<<<(nw + 255) / 256, 256, 0, stream>>>(qkv_w, w16, nw);

  win_attn<4, 0><<<4096, 256, 0, stream>>>(x, w16, qkv_b, tab0, out);
  win_attn<8, 1><<<1024, 256, 0, stream>>>(x, w16, qkv_b, tab1, out);
}
